// WaveRNN_13640816132171
// MI455X (gfx1250) — matrix operand order repaired and checked by reading, not compiled, not run on hardware
//
#include <hip/hip_runtime.h>
#include <cstdint>
#include <cstddef>

// ---------------------------------------------------------------------------
// Types / helpers
// ---------------------------------------------------------------------------
typedef __bf16 v16bf __attribute__((ext_vector_type(16)));
typedef float  v8f   __attribute__((ext_vector_type(8)));

union BFVec { v16bf v; unsigned short s[16]; };

__device__ __forceinline__ unsigned short f32_to_bf16(float f) {
  unsigned u = __builtin_bit_cast(unsigned, f);
  u += 0x7FFFu + ((u >> 16) & 1u);          // round-to-nearest-even
  return (unsigned short)(u >> 16);
}
__device__ __forceinline__ float bf16_to_f32(unsigned short h) {
  unsigned u = ((unsigned)h) << 16;
  return __builtin_bit_cast(float, u);
}

// Fragment layouts used here (contiguous k, not the ISA table order), 16-bit data, wave32:
//  A (16x32, M x K):  lane: m = lane&15, lh = lane>>4; half i -> v=i>>1,e=i&1,
//                     K = lh*16 + i
//  B (32x16, K x N):  lane: n = lane&15, kb = (lane>>4)*16; half i -> K = kb + i
// We store fragments as 32-byte blobs: [entry][lane][16 halfs], so loads are
// two ds_load_b128 per fragment.

// ---------------------------------------------------------------------------
// Mel upsample stage: repeat(scale) + 'SAME' box FIR (taps = 2S+1)
// layout (B, T, M) row-major
// ---------------------------------------------------------------------------
__global__ void mel_stage_kernel(const float* __restrict__ in, float* __restrict__ out,
                                 const float* __restrict__ w,
                                 int B, int Tin, int M, int scale, int S) {
  int idx = blockIdx.x * blockDim.x + threadIdx.x;
  int Tout = Tin * scale;
  int total = B * Tout * M;
  if (idx >= total) return;
  int m = idx % M;
  int t = (idx / M) % Tout;
  int b = idx / (M * Tout);
  float acc = 0.f;
  for (int d = -S; d <= S; ++d) {
    int tt = t + d;
    if (tt >= 0 && tt < Tout) acc += w[d + S] * in[((size_t)b * Tin + (tt / scale)) * M + m];
  }
  out[idx] = acc;
}

// ---------------------------------------------------------------------------
// Aux (melresnet) network: one block per (b,t) row, 128 threads = channels
// ---------------------------------------------------------------------------
__global__ __launch_bounds__(128) void aux_kernel(
    const float* __restrict__ mels, const float* __restrict__ w0,
    const float* __restrict__ g0, const float* __restrict__ b0,
    const float* __restrict__ m0, const float* __restrict__ v0,
    const float* __restrict__ w1, const float* __restrict__ g1,
    const float* __restrict__ b1, const float* __restrict__ m1,
    const float* __restrict__ v1, const float* __restrict__ w2,
    const float* __restrict__ g2, const float* __restrict__ b2,
    const float* __restrict__ m2, const float* __restrict__ v2,
    const float* __restrict__ wout, const float* __restrict__ bout,
    float* __restrict__ aout) {
  const int bt = blockIdx.x;         // 0..159
  const int b = bt / 40, t = bt % 40;
  const int c = threadIdx.x;         // 0..127
  __shared__ float sa[128], sy[128];
  float acc = 0.f;
  for (int dt = 0; dt < 5; ++dt)
    for (int m = 0; m < 80; ++m)
      acc += mels[((size_t)b * 44 + t + dt) * 80 + m] * w0[((size_t)dt * 80 + m) * 128 + c];
  float x = (acc - m0[c]) * rsqrtf(v0[c] + 1e-3f) * g0[c] + b0[c];
  sa[c] = fmaxf(x, 0.f);
  __syncthreads();
  for (int i = 0; i < 10; ++i) {
    const float* W1 = w1 + (size_t)i * 128 * 128;
    const float* W2 = w2 + (size_t)i * 128 * 128;
    float a1 = 0.f;
    for (int k = 0; k < 128; ++k) a1 += sa[k] * W1[k * 128 + c];
    float y = (a1 - m1[i * 128 + c]) * rsqrtf(v1[i * 128 + c] + 1e-3f) * g1[i * 128 + c] + b1[i * 128 + c];
    sy[c] = fmaxf(y, 0.f);
    __syncthreads();
    float a2 = 0.f;
    for (int k = 0; k < 128; ++k) a2 += sy[k] * W2[k * 128 + c];
    float y2 = (a2 - m2[i * 128 + c]) * rsqrtf(v2[i * 128 + c] + 1e-3f) * g2[i * 128 + c] + b2[i * 128 + c];
    __syncthreads();
    sa[c] = sa[c] + y2;
    __syncthreads();
  }
  float o = bout[c];
  for (int k = 0; k < 128; ++k) o += sa[k] * wout[k * 128 + c];
  aout[(size_t)bt * 128 + c] = o;
}

// ---------------------------------------------------------------------------
// Pack Xin (40960 x 128 f32): [audio | mels_up(80) | a0(32) | zero-pad]
// ---------------------------------------------------------------------------
__global__ void pack_xin_kernel(const float* __restrict__ audios,
                                const float* __restrict__ U, // (B,11264,80)
                                const float* __restrict__ aaux,
                                float* __restrict__ Xin) {
  const size_t total = 40960ull * 128ull;
  for (size_t idx = (size_t)blockIdx.x * blockDim.x + threadIdx.x; idx < total;
       idx += (size_t)gridDim.x * blockDim.x) {
    int col = (int)(idx & 127);
    size_t row = idx >> 7;
    int b = (int)(row / 10240), t = (int)(row % 10240);
    float v = 0.f;
    if (col == 0)       v = audios[(size_t)b * 10241 + t];
    else if (col < 81)  v = U[((size_t)b * 11264 + t + 512) * 80 + (col - 1)];
    else if (col < 113) v = aaux[((size_t)b * 40 + (t >> 8)) * 128 + (col - 81)];
    Xin[idx] = v;
  }
}

// ---------------------------------------------------------------------------
// Weight f32 -> bf16 in WMMA B-fragment order:
// dst layout [Kd/32][Nd/16][lane(32)][16 halfs]; zero-padded outside (Ks,Ns)
// ---------------------------------------------------------------------------
__global__ void wconv_frag_kernel(const float* __restrict__ src, unsigned short* __restrict__ dst,
                                  int Ks, int Ns, int Kd, int Nd) {
  const int ntiles = Nd >> 4;
  const size_t total = (size_t)(Kd >> 5) * ntiles * 32;   // fragment entries
  for (size_t eidx = (size_t)blockIdx.x * blockDim.x + threadIdx.x; eidx < total;
       eidx += (size_t)gridDim.x * blockDim.x) {
    int lane = (int)(eidx & 31);
    size_t r = eidx >> 5;
    int ntg = (int)(r % ntiles);
    int ks = (int)(r / ntiles);
    int n = ntg * 16 + (lane & 15);
    int kb = ks * 32 + ((lane >> 4) * 16);
    unsigned short* d = dst + eidx * 16;
#pragma unroll
    for (int i = 0; i < 16; ++i) {
      int k = kb + i;
      float f = (k < Ks && n < Ns) ? src[(size_t)k * Ns + n] : 0.f;
      d[i] = f32_to_bf16(f);
    }
  }
}

// ---------------------------------------------------------------------------
// Generic bf16 WMMA GEMM: C = act( [A | aux] * B + bias ) (+res)
//  - A f32, converted bf16 while staging directly in A-fragment order
//  - B pre-swizzled global fragment buffer, coalesced uint4 copy into LDS
//  - inner loop: 1 vector LDS A-load + 4 vector LDS B-loads + 4 wmma
// Block: 256 thr (8 waves), tile 128(M) x 64(N), K-step 32
// ---------------------------------------------------------------------------
#define GB_BM 128
#define GB_BN 64
__global__ __launch_bounds__(256) void gemm_bf16_kernel(
    const float* __restrict__ A, int lda,
    const float* __restrict__ aux, int auxoff, int K1, int K2,
    const unsigned short* __restrict__ BwFrag, int ntilesB,
    const float* __restrict__ bias,
    const float* __restrict__ res, int ldres,
    float* __restrict__ Cf, int ldc,
    unsigned short* __restrict__ Cbf, int ldcbf,
    int M, int N, int Nstore, int relu, int T) {
  __shared__ unsigned short sA[8 * 32 * 16];   // 8 KB, fragment order [wave][lane][16]
  __shared__ unsigned short sB[4 * 32 * 16];   // 4 KB, fragment order [ntile][lane][16]
  const int tid = threadIdx.x, lane = tid & 31, w = tid >> 5;
  const int bm = blockIdx.x * GB_BM, bn = blockIdx.y * GB_BN;
  const int bt16 = bn >> 4;
  const int K = K1 + K2;
  const v8f vz = {0.f, 0.f, 0.f, 0.f, 0.f, 0.f, 0.f, 0.f};
  v8f acc[4];
  acc[0] = vz; acc[1] = vz; acc[2] = vz; acc[3] = vz;

  // This thread's fixed A-fragment entry (wave = tid>>5, lane = tid&31)
  const int mrow = bm + (tid >> 5) * 16 + (lane & 15);
  const int lh = (lane >> 4) & 1;
  const float* arow = &A[(size_t)mrow * lda];
  const float* auxrow = nullptr;
  if (aux) {
    int b = mrow / T, tt = mrow - b * T;
    auxrow = &aux[((size_t)b * 40 + (tt >> 8)) * 128 + auxoff];
  }
  // B copy assignment: thread -> (ntile = tid>>6, q = tid&63) uint4 within tile
  const int ntc = tid >> 6, qc = tid & 63;

  for (int k0 = 0, ks = 0; k0 < K; k0 += 32, ++ks) {
    // stage A fragment entry (16 strided f32 reads -> one 32B LDS store)
    BFVec ua;
#pragma unroll
    for (int i = 0; i < 16; ++i) {
      int v = i >> 1, e = i & 1;
      int k = k0 + lh * 16 + i;
      float f = (k < K1) ? arow[k] : auxrow[k - K1];
      ua.s[i] = f32_to_bf16(f);
    }
    *(v16bf*)(void*)&sA[tid * 16] = ua.v;
    // stage B: 4 n-tiles x 1 KB, coalesced 16B copies
    {
      const uint4* srcq =
          (const uint4*)(const void*)&BwFrag[(((size_t)ks * ntilesB + bt16 + ntc) * 32) * 16];
      uint4* dstq = (uint4*)(void*)&sB[(ntc * 32) * 16];
      dstq[qc] = srcq[qc];
    }
    __syncthreads();
    v16bf af = *(const v16bf*)(const void*)&sA[(w * 32 + lane) * 16];
#pragma unroll
    for (int nt = 0; nt < 4; ++nt) {
      v16bf bfr = *(const v16bf*)(const void*)&sB[(nt * 32 + lane) * 16];
      acc[nt] = __builtin_amdgcn_wmma_f32_16x16x32_bf16(false, af, false, bfr,
                                                        (short)0, acc[nt], false, false);
    }
    __syncthreads();
  }
  const int nl = lane & 15, mh = (lane >> 4) * 8;
  for (int nt = 0; nt < 4; ++nt) {
#pragma unroll
    for (int v = 0; v < 8; ++v) {
      int row = bm + w * 16 + mh + v;
      int col = bn + nt * 16 + nl;
      if (row < M && col < Nstore) {
        float val = acc[nt][v];
        if (bias) val += bias[col];
        if (relu) val = fmaxf(val, 0.f);
        if (res)  val += res[(size_t)row * ldres + col];
        if (Cf)   Cf[(size_t)row * ldc + col] = val;
        if (Cbf)  Cbf[(size_t)row * ldcbf + col] = f32_to_bf16(val);
      }
    }
  }
}

// ---------------------------------------------------------------------------
// Persistent GRU: 16 cooperating workgroups, WG g owns h[32g..32g+32).
// Wh slice (512 x 96 cols: z|r|hh for this slice) lives in LDS in WMMA
// B-fragment order (96 KB). Per step: 6 waves x 16 x v_wmma_f32_16x16x32_bf16,
// sigmoid/tanh update, device-wide barrier (s_cluster_barrier fast path +
// atomic sense-reversing barrier).
// ---------------------------------------------------------------------------
__global__ __launch_bounds__(256) void gru_kernel(
    const unsigned short* __restrict__ gx,   // (B*T, 1536) bf16 x-projection (incl bx)
    const float* __restrict__ Wh,            // (512, 1536) f32
    const float* __restrict__ bh,            // (1536)
    const float* __restrict__ res,           // (B*T, 512) residual input
    float* __restrict__ y,                   // (B*T, 512) output
    float* __restrict__ h0, float* __restrict__ h1,  // (4 x 512) double buffer
    unsigned* barCnt, unsigned* barGen, int T) {
  __shared__ unsigned short sWh[6 * 16 * 32 * 16];  // 96 KB, B-fragment order
  __shared__ unsigned short sA[16 * 32 * 16];       // 16 KB, A-fragment order
  __shared__ float sRP[96 * 16];                    // h@Wh partials [col][m]
  const int g = blockIdx.x;                          // 0..15
  const int tid = threadIdx.x, lane = tid & 31, w = tid >> 5;

  // Pre-swizzle this WG's Wh column slice into fragment order (bf16)
  for (int idx = tid; idx < 6 * 16 * 32; idx += 256) {
    int tile = idx / 512;
    int rem = idx - tile * 512;
    int kstep = rem >> 5;
    int ln = rem & 31;
    int gate = tile >> 1;
    int col = gate * 512 + g * 32 + (tile & 1) * 16 + (ln & 15);
    int kb = kstep * 32 + ((ln >> 4) * 16);
    BFVec uw;
#pragma unroll
    for (int i = 0; i < 16; ++i)
      uw.s[i] = f32_to_bf16(Wh[(size_t)(kb + i) * 1536 + col]);
    *(v16bf*)(void*)&sWh[idx * 16] = uw.v;
  }
  __syncthreads();

  for (int t = 0; t < T; ++t) {
    const float* hc = (t & 1) ? h1 : h0;
    float*       hn = (t & 1) ? h0 : h1;
    // Build A fragments: rows 0..3 = batch hidden state (bf16), rows 4..15 = 0
    for (int idx = tid; idx < 512; idx += 256) {
      int kstep = idx >> 5, ln = idx & 31;
      int m = ln & 15, lh = (ln >> 4) & 1;
      BFVec ua;
#pragma unroll
      for (int i = 0; i < 16; ++i) {
        int v = i >> 1, e = i & 1;
        int k = kstep * 32 + lh * 16 + i;
        ua.s[i] = (m < 4) ? f32_to_bf16(hc[m * 512 + k]) : (unsigned short)0;
      }
      *(v16bf*)(void*)&sA[idx * 16] = ua.v;
    }
    __syncthreads();
    if (w < 6) {
      v8f acc = {0.f, 0.f, 0.f, 0.f, 0.f, 0.f, 0.f, 0.f};
#pragma unroll
      for (int ks = 0; ks < 16; ++ks) {
        v16bf a   = *(const v16bf*)(const void*)&sA[(ks * 32 + lane) * 16];
        v16bf bfr = *(const v16bf*)(const void*)&sWh[((w * 16 + ks) * 32 + lane) * 16];
        acc = __builtin_amdgcn_wmma_f32_16x16x32_bf16(false, a, false, bfr,
                                                      (short)0, acc, false, false);
      }
      int n = lane & 15, mh = (lane >> 4) * 8;
#pragma unroll
      for (int v = 0; v < 8; ++v) sRP[(w * 16 + n) * 16 + mh + v] = acc[v];
    }
    // Prefetch next step's gate pre-activations while WMMA drains
    if (tid < 128 && (t + 1) < T) {
      int b = tid >> 5, jo = tid & 31;
      __builtin_prefetch(&gx[(size_t)(b * T + t + 1) * 1536 + g * 32 + jo], 0, 1);
    }
    __syncthreads();
    if (tid < 128) {
      int b = tid >> 5, jo = tid & 31;
      int j = g * 32 + jo;
      size_t ro = (size_t)(b * T + t) * 1536;
      float xz = bf16_to_f32(gx[ro + j]);
      float xr = bf16_to_f32(gx[ro + 512 + j]);
      float xh = bf16_to_f32(gx[ro + 1024 + j]);
      float rz = sRP[jo * 16 + b]        + bh[j];
      float rr = sRP[(32 + jo) * 16 + b] + bh[512 + j];
      float rh = sRP[(64 + jo) * 16 + b] + bh[1024 + j];
      float z = 1.f / (1.f + __expf(-(xz + rz)));
      float r = 1.f / (1.f + __expf(-(xr + rr)));
      float hh = tanhf(xh + r * rh);
      float hp = hc[b * 512 + j];
      float hv = z * hp + (1.f - z) * hh;
      hn[b * 512 + j] = hv;
      size_t yo = (size_t)(b * T + t) * 512 + j;
      y[yo] = hv + res[yo];
    }
    // Device-wide barrier across 16 WGs
    __syncthreads();
    if (tid == 0) {
      __builtin_amdgcn_s_cluster_barrier();  // HW fast path when clustered; NOP otherwise
      __threadfence();
      unsigned gen0 = __hip_atomic_load(barGen, __ATOMIC_ACQUIRE, __HIP_MEMORY_SCOPE_AGENT);
      unsigned a = __hip_atomic_fetch_add(barCnt, 1u, __ATOMIC_ACQ_REL, __HIP_MEMORY_SCOPE_AGENT);
      if (a == 15u) {
        __hip_atomic_store(barCnt, 0u, __ATOMIC_RELAXED, __HIP_MEMORY_SCOPE_AGENT);
        __hip_atomic_fetch_add(barGen, 1u, __ATOMIC_RELEASE, __HIP_MEMORY_SCOPE_AGENT);
      } else {
        while (__hip_atomic_load(barGen, __ATOMIC_ACQUIRE, __HIP_MEMORY_SCOPE_AGENT) == gen0)
          __builtin_amdgcn_s_sleep(1);
      }
    }
    __syncthreads();
  }
}

// ---------------------------------------------------------------------------
// Host launcher
// ---------------------------------------------------------------------------
extern "C" void kernel_launch(void* const* d_in, const int* in_sizes, int n_in,
                              void* d_out, int out_size, void* d_ws, size_t ws_size,
                              hipStream_t stream) {
  (void)in_sizes; (void)n_in; (void)out_size; (void)ws_size;
  const float* audios = (const float*)d_in[0];
  const float* mels   = (const float*)d_in[1];
  const float* wm0    = (const float*)d_in[2];
  const float* wm1    = (const float*)d_in[3];
  const float* wm2    = (const float*)d_in[4];
  const float* cw0    = (const float*)d_in[5];
  const float* bn0g   = (const float*)d_in[6];
  const float* bn0b   = (const float*)d_in[7];
  const float* bn0m   = (const float*)d_in[8];
  const float* bn0v   = (const float*)d_in[9];
  const float* rw1    = (const float*)d_in[10];
  const float* bn1g   = (const float*)d_in[11];
  const float* bn1b   = (const float*)d_in[12];
  const float* bn1m   = (const float*)d_in[13];
  const float* bn1v   = (const float*)d_in[14];
  const float* rw2    = (const float*)d_in[15];
  const float* bn2g   = (const float*)d_in[16];
  const float* bn2b   = (const float*)d_in[17];
  const float* bn2m   = (const float*)d_in[18];
  const float* bn2v   = (const float*)d_in[19];
  const float* wOut   = (const float*)d_in[20];
  const float* bOut   = (const float*)d_in[21];
  const float* Wi     = (const float*)d_in[22];
  const float* bi     = (const float*)d_in[23];
  const float* Wx0    = (const float*)d_in[24];
  const float* Wh0    = (const float*)d_in[25];
  const float* bx0    = (const float*)d_in[26];
  const float* bh0    = (const float*)d_in[27];
  const float* Wx1    = (const float*)d_in[28];
  const float* Wh1    = (const float*)d_in[29];
  const float* bx1    = (const float*)d_in[30];
  const float* bh1    = (const float*)d_in[31];
  const float* Wd0    = (const float*)d_in[32];
  const float* bd0    = (const float*)d_in[33];
  const float* Wd1    = (const float*)d_in[34];
  const float* bd1    = (const float*)d_in[35];
  const float* Wp     = (const float*)d_in[36];
  const float* bp     = (const float*)d_in[37];

  // --- scratch bump allocator (256B aligned) ---
  char* ws = (char*)d_ws;
  size_t off = 0;
  auto alloc = [&](size_t bytes) -> char* {
    char* p = ws + off;
    off += (bytes + 255) & ~(size_t)255;
    return p;
  };
  float* U0   = (float*)alloc(4ull * 11264 * 80 * 4);
  float* U1   = (float*)alloc(4ull * 11264 * 80 * 4);
  float* Aaux = (float*)alloc(160ull * 128 * 4);
  float* Xin  = (float*)alloc(40960ull * 128 * 4);
  float* S0   = (float*)alloc(40960ull * 512 * 4);
  float* S1   = (float*)alloc(40960ull * 512 * 4);
  float* S2   = (float*)alloc(40960ull * 512 * 4);
  unsigned short* GX   = (unsigned short*)alloc(40960ull * 1536 * 2);
  unsigned short* WiB  = (unsigned short*)alloc(128ull * 512 * 2);
  unsigned short* Wx0B = (unsigned short*)alloc(512ull * 1536 * 2);
  unsigned short* Wx1B = (unsigned short*)alloc(544ull * 1536 * 2);
  unsigned short* Wd0B = (unsigned short*)alloc(544ull * 512 * 2);
  unsigned short* Wd1B = (unsigned short*)alloc(544ull * 512 * 2);
  unsigned short* WpB  = (unsigned short*)alloc(512ull * 64 * 2);   // N padded to 64
  float* H0 = (float*)alloc(4ull * 512 * 4);
  float* H1 = (float*)alloc(4ull * 512 * 4);
  unsigned* Bar = (unsigned*)alloc(256);

  const int T = 10240;
  dim3 blk(256);

  // --- weight conversion to padded bf16 fragment buffers ---
  auto wc = [&](const float* s, unsigned short* d, int Ks, int Ns, int Kd, int Nd) {
    int total = (Kd / 32) * (Nd / 16) * 32;     // fragment entries
    wconv_frag_kernel<<<dim3((total + 255) / 256), blk, 0, stream>>>(s, d, Ks, Ns, Kd, Nd);
  };
  wc(Wi,  WiB,  113, 512,  128, 512);
  wc(Wx0, Wx0B, 512, 1536, 512, 1536);
  wc(Wx1, Wx1B, 544, 1536, 544, 1536);
  wc(Wd0, Wd0B, 544, 512,  544, 512);
  wc(Wd1, Wd1B, 544, 512,  544, 512);
  wc(Wp,  WpB,  512, 30,   512, 64);

  // --- mel upsampling: 44 -> 176 -> 1408 -> 11264 ---
  mel_stage_kernel<<<dim3((4 * 176 * 80 + 255) / 256), blk, 0, stream>>>(mels, U0, wm0, 4, 44, 80, 4, 4);
  mel_stage_kernel<<<dim3((4 * 1408 * 80 + 255) / 256), blk, 0, stream>>>(U0, U1, wm1, 4, 176, 80, 8, 8);
  mel_stage_kernel<<<dim3((4 * 11264 * 80 + 255) / 256), blk, 0, stream>>>(U1, U0, wm2, 4, 1408, 80, 8, 8);

  // --- aux network ---
  aux_kernel<<<dim3(160), dim3(128), 0, stream>>>(mels, cw0, bn0g, bn0b, bn0m, bn0v,
                                                  rw1, bn1g, bn1b, bn1m, bn1v,
                                                  rw2, bn2g, bn2b, bn2m, bn2v,
                                                  wOut, bOut, Aaux);

  // --- pack GRU-input features ---
  pack_xin_kernel<<<dim3(20480), blk, 0, stream>>>(audios, U0, Aaux, Xin);

  // --- GEMM helper ---
  auto gemm = [&](const float* A, int lda, const float* aux, int auxoff, int K1, int K2,
                  const unsigned short* BwFrag, int Npad, const float* bias,
                  const float* res, float* Cf, int ldc,
                  unsigned short* Cbf, int ldcbf, int N, int Nstore, int relu) {
    dim3 grid(40960 / GB_BM, (unsigned)((N + GB_BN - 1) / GB_BN));
    gemm_bf16_kernel<<<grid, blk, 0, stream>>>(A, lda, aux, auxoff, K1, K2, BwFrag, Npad / 16,
                                               bias, res, 512, Cf, ldc, Cbf, ldcbf,
                                               40960, N, Nstore, relu, T);
  };

  // G1: X0 = Xin @ Wi + bi                     -> S0 (f32)
  gemm(Xin, 128, nullptr, 0, 128, 0, WiB, 512, bi, nullptr, S0, 512, nullptr, 0, 512, 512, 0);
  // G2: gx0 = X0 @ Wx0 + bx0                   -> GX (bf16)
  gemm(S0, 512, nullptr, 0, 512, 0, Wx0B, 1536, bx0, nullptr, nullptr, 0, GX, 1536, 1536, 1536, 0);

  // GRU0: y0 = gru(gx0; Wh0, bh0) + X0         -> S1
  hipMemsetAsync(H0, 0, 4 * 512 * sizeof(float), stream);
  hipMemsetAsync(H1, 0, 4 * 512 * sizeof(float), stream);
  hipMemsetAsync(Bar, 0, 256, stream);
  gru_kernel<<<dim3(16), blk, 0, stream>>>(GX, Wh0, bh0, S0, S1, H0, H1, Bar, Bar + 1, T);

  // G3: gx1 = [y0 | a1] @ Wx1 + bx1            -> GX (bf16)
  gemm(S1, 512, Aaux, 32, 512, 32, Wx1B, 1536, bx1, nullptr, nullptr, 0, GX, 1536, 1536, 1536, 0);

  // GRU1: y1 = gru(gx1; Wh1, bh1) + y0         -> S2
  hipMemsetAsync(H0, 0, 4 * 512 * sizeof(float), stream);
  hipMemsetAsync(H1, 0, 4 * 512 * sizeof(float), stream);
  hipMemsetAsync(Bar, 0, 256, stream);
  gru_kernel<<<dim3(16), blk, 0, stream>>>(GX, Wh1, bh1, S1, S2, H0, H1, Bar, Bar + 1, T);

  // G4: y2 = relu([y1 | a2] @ Wd0 + bd0) + y1  -> S0
  gemm(S2, 512, Aaux, 64, 512, 32, Wd0B, 512, bd0, S2, S0, 512, nullptr, 0, 512, 512, 1);
  // G5: y3 = relu([y2 | a3] @ Wd1 + bd1) + y2  -> S1
  gemm(S0, 512, Aaux, 96, 512, 32, Wd1B, 512, bd1, S0, S1, 512, nullptr, 0, 512, 512, 1);
  // G6: logits = y3 @ Wp + bp                  -> d_out (N padded to 64, store 30)
  gemm(S1, 512, nullptr, 0, 512, 0, WpB, 64, bp, nullptr, (float*)d_out, 30, nullptr, 0, 32, 30, 0);
}
